// LocalTransformer_55748675502259
// MI455X (gfx1250) — matrix operand order repaired and checked by reading, not compiled, not run on hardware
//
#include <hip/hip_runtime.h>
#include <hip/hip_bf16.h>
#include <cstddef>

// ---------------------------------------------------------------------------
// Problem constants (match reference)
// ---------------------------------------------------------------------------
#define BB   2
#define NN   16384
#define CC   256
#define NPP  2048
#define NSS  32
#define HH   4
#define LL   2
#define DFFF 512
#define GG   (BB * NPP)          // 4096 groups
#define HD   (CC / HH)           // 64
#define LF   (NPP * NSS)         // 65536

typedef __attribute__((ext_vector_type(16))) _Float16 v16h;
typedef __attribute__((ext_vector_type(8)))  float    v8f;

// ---------------------------------------------------------------------------
// WMMA fragment helpers (gfx1250 wave32 layouts, CDNA5 ISA)
// ---------------------------------------------------------------------------
// A: 16x32 f16.  lane l: m = l&15; koff = (l>>4)*8; halves 0..7 -> k=koff+i,
// halves 8..15 -> k=16+koff+i.  base points at (m0,k0), ld = row stride.
__device__ __forceinline__ v16h lt_frag_a(const _Float16* base, int ld, int lane) {
    const _Float16* p = base + (lane & 15) * ld + ((lane >> 4) << 3);
    v16h f;
#pragma unroll
    for (int i = 0; i < 8; ++i) { f[i] = p[i]; f[i + 8] = p[i + 16]; }
    return f;
}

// B: 32x16 f16 where B[k][n] is stored as row-major [N][K] (i.e. weight W[n][k]).
// lane l: n = l&15; koff = (l>>4)*8; halves 0..7 -> k=koff+i, halves 8..15 -> k=16+koff+i.
__device__ __forceinline__ v16h lt_frag_b(const _Float16* base, int ld, int lane) {
    const _Float16* p = base + (lane & 15) * ld + ((lane >> 4) << 3);
    v16h f;
#pragma unroll
    for (int i = 0; i < 8; ++i) { f[i] = p[i]; f[i + 8] = p[i + 16]; }
    return f;
}

__device__ __forceinline__ v8f lt_wmma(v16h a, v16h b, v8f c) {
    return __builtin_amdgcn_wmma_f32_16x16x32_f16(
        /*neg_a=*/false, a, /*neg_b=*/false, b,
        /*c_mod=*/(short)0, c, /*reuse_a=*/false, /*reuse_b=*/false);
}

// ---------------------------------------------------------------------------
// f32 -> f16 weight conversion
// ---------------------------------------------------------------------------
__global__ void lt_cvt_f16_kernel(const float* __restrict__ src,
                                  _Float16* __restrict__ dst, int n) {
    int i = blockIdx.x * 256 + threadIdx.x;
    if (i < n) dst[i] = (_Float16)src[i];
}

// ---------------------------------------------------------------------------
// Farthest point sampling: one block per batch, iterative scan.
// ---------------------------------------------------------------------------
__global__ __launch_bounds__(256)
void lt_fps_kernel(const float* __restrict__ xyz, float* __restrict__ distbuf,
                   int* __restrict__ fpsidx) {
    __shared__ float sv[256];
    __shared__ int   si[256];
    const int b = blockIdx.x;
    const int tid = threadIdx.x;
    float* dist = distbuf + (size_t)b * NN;
    const float* xb = xyz + (size_t)b * NN * 3;

    for (int i = tid; i < NN; i += 256) dist[i] = 1e10f;
    __syncthreads();

    int far = 0;
    for (int it = 0; it < NPP; ++it) {
        if (tid == 0) fpsidx[b * NPP + it] = far;
        const float cx = xb[far * 3 + 0];
        const float cy = xb[far * 3 + 1];
        const float cz = xb[far * 3 + 2];
        float best = -1.0f; int bi = 0;
        for (int i = tid; i < NN; i += 256) {
            const float dx = xb[i * 3 + 0] - cx;
            const float dy = xb[i * 3 + 1] - cy;
            const float dz = xb[i * 3 + 2] - cz;
            float d = fminf(dist[i], dx * dx + dy * dy + dz * dz);
            dist[i] = d;
            if (d > best) { best = d; bi = i; }
        }
        sv[tid] = best; si[tid] = bi;
        __syncthreads();
        for (int s = 128; s > 0; s >>= 1) {
            if (tid < s && sv[tid + s] > sv[tid]) { sv[tid] = sv[tid + s]; si[tid] = si[tid + s]; }
            __syncthreads();
        }
        far = si[0];
        __syncthreads();
    }
}

// ---------------------------------------------------------------------------
// Ball query: first <=NS indices (ascending) within radius, padded w/ first.
// ---------------------------------------------------------------------------
__global__ __launch_bounds__(256)
void lt_ball_query_kernel(const float* __restrict__ xyz,
                          const int* __restrict__ fpsidx,
                          int* __restrict__ idxbuf) {
    const int ci = blockIdx.x * 256 + threadIdx.x;
    if (ci >= GG) return;
    const int b = ci / NPP;
    const int center = fpsidx[ci];
    const float* xb = xyz + (size_t)b * NN * 3;
    const float cx = xb[center * 3 + 0];
    const float cy = xb[center * 3 + 1];
    const float cz = xb[center * 3 + 2];
    int* row = idxbuf + (size_t)ci * NSS;
    int cnt = 0;
    for (int j = 0; j < NN && cnt < NSS; ++j) {
        const float dx = xb[j * 3 + 0] - cx;
        const float dy = xb[j * 3 + 1] - cy;
        const float dz = xb[j * 3 + 2] - cz;
        if (dx * dx + dy * dy + dz * dz < 1.0f) row[cnt++] = j;
    }
    const int pad = (cnt > 0) ? row[0] : 0;
    for (; cnt < NSS; ++cnt) row[cnt] = pad;
}

// ---------------------------------------------------------------------------
// Gather + positional encoding: x0[g][s][c] = feat[b][c][idx] + PE(g_xyz)
// ---------------------------------------------------------------------------
__global__ __launch_bounds__(256)
void lt_pe_gather_kernel(const float* __restrict__ xyz, const float* __restrict__ feat,
                         const float* __restrict__ pe1w, const float* __restrict__ pe1b,
                         const float* __restrict__ bng, const float* __restrict__ bnb,
                         const float* __restrict__ bnm, const float* __restrict__ bnv,
                         const float* __restrict__ pe2w, const float* __restrict__ pe2b,
                         const int* __restrict__ fpsidx, const int* __restrict__ idxbuf,
                         float* __restrict__ xbuf) {
    __shared__ float hS[128];
    const int g = blockIdx.x;
    const int b = g / NPP;
    const int tid = threadIdx.x;
    const int center = fpsidx[g];
    const float* xb = xyz + (size_t)b * NN * 3;
    const float cx = xb[center * 3 + 0];
    const float cy = xb[center * 3 + 1];
    const float cz = xb[center * 3 + 2];
    const int* row = idxbuf + (size_t)g * NSS;
    float* xg = xbuf + (size_t)g * NSS * CC;

    for (int s = 0; s < NSS; ++s) {
        const int j = row[s];
        const float dx = xb[j * 3 + 0] - cx;
        const float dy = xb[j * 3 + 1] - cy;
        const float dz = xb[j * 3 + 2] - cz;
        if (tid < 128) {
            float hp = pe1w[tid * 3 + 0] * dx + pe1w[tid * 3 + 1] * dy +
                       pe1w[tid * 3 + 2] * dz + pe1b[tid];
            float hn = (hp - bnm[tid]) * rsqrtf(bnv[tid] + 1e-5f);
            hS[tid] = fmaxf(hn * bng[tid] + bnb[tid], 0.0f);
        }
        __syncthreads();
        const int c = tid;
        float acc = pe2b[c];
        const float* w = pe2w + c * 128;
#pragma unroll 8
        for (int q = 0; q < 128; ++q) acc += w[q] * hS[q];
        xg[s * CC + c] = feat[((size_t)b * CC + c) * NN + j] + acc;
        __syncthreads();
    }
}

// ---------------------------------------------------------------------------
// LayerNorm over C (8 lanes per token, shfl reduce), write f16 A-operand.
// ---------------------------------------------------------------------------
__device__ __forceinline__ void lt_ln(const float* __restrict__ xS,
                                      _Float16* __restrict__ aS,
                                      const float* __restrict__ sc,
                                      const float* __restrict__ bi, int tid) {
    const int t = tid >> 3, sub = tid & 7;
    const float* rowp = xS + t * CC;
    float s1 = 0.f, s2 = 0.f;
#pragma unroll 4
    for (int i = 0; i < 32; ++i) {
        float v = rowp[sub * 32 + i];
        s1 += v; s2 += v * v;
    }
#pragma unroll
    for (int m = 1; m < 8; m <<= 1) {
        s1 += __shfl_xor(s1, m, 32);
        s2 += __shfl_xor(s2, m, 32);
    }
    const float mean = s1 * (1.0f / 256.0f);
    const float var  = s2 * (1.0f / 256.0f) - mean * mean;
    const float rstd = rsqrtf(var + 1e-5f);
    _Float16* arow = aS + t * CC;
    for (int i = 0; i < 32; ++i) {
        const int c = sub * 32 + i;
        arow[c] = (_Float16)((rowp[c] - mean) * rstd * sc[c] + bi[c]);
    }
}

// ---------------------------------------------------------------------------
// Fused 2-layer local transformer: one workgroup (8 wave32) per group.
// Residual stream x [32x256] stays in LDS for both layers; all GEMMs via
// v_wmma_f32_16x16x32_f16 with register-cached A fragments.
// 120 KB dynamic LDS -> 2 groups per 320KB WGP.
// ---------------------------------------------------------------------------
__global__ __launch_bounds__(256)
void lt_transformer_kernel(float* __restrict__ xbuf,
                           const _Float16* __restrict__ wqkv, const float* __restrict__ bqkv,
                           const _Float16* __restrict__ wo,   const float* __restrict__ bo,
                           const float* __restrict__ ln1s, const float* __restrict__ ln1b,
                           const float* __restrict__ ln2s, const float* __restrict__ ln2b,
                           const _Float16* __restrict__ wfc1, const float* __restrict__ bfc1,
                           const _Float16* __restrict__ wfc2, const float* __restrict__ bfc2) {
    extern __shared__ char smem[];
    float*    xS   = (float*)smem;                       // 32*256 f32  [0,32K)
    _Float16* aS   = (_Float16*)(smem + 32 * 1024);      // 32*256 f16  [32K,48K)
    _Float16* qS   = (_Float16*)(smem + 48 * 1024);      // 32*256 f16 [token][chan]
    _Float16* kS   = qS + 32 * 256;                      // 32*256 f16 [token][chan]
    _Float16* vT   = kS + 32 * 256;                      // 256*32 f16 [chan][token]
    float*    attF = (float*)(smem + 96 * 1024);         // 4*32*32 f32 [96K,112K)
    _Float16* attH = (_Float16*)(smem + 112 * 1024);     // 4*32*32 f16 [112K,120K)
    _Float16* hS   = qS;                                 // 32*512 f16 (reuse q/k)

    const int g = blockIdx.x;
    const int tid = threadIdx.x;
    const int lane = tid & 31;
    const int wv = tid >> 5;
    const int hi8 = (lane >> 4) << 3;       // 0 or 8 (C/D row offset)
    float* xg = xbuf + (size_t)g * (NSS * CC);

    {
        const float4* src4 = (const float4*)xg;
        float4* dst4 = (float4*)xS;
        for (int i = tid; i < (NSS * CC) / 4; i += 256) dst4[i] = src4[i];
    }
    __syncthreads();

    for (int l = 0; l < LL; ++l) {
        const _Float16* Wqkv = wqkv + (size_t)l * 768 * 256;
        const _Float16* Wo   = wo   + (size_t)l * 256 * 256;
        const _Float16* Wf1  = wfc1 + (size_t)l * 512 * 256;
        const _Float16* Wf2  = wfc2 + (size_t)l * 256 * 512;

        // ---- LN1 -> aS (f16) ----
        lt_ln(xS, aS, ln1s + l * CC, ln1b + l * CC, tid);
        __syncthreads();

        // ---- QKV: [32x256] @ W^T[256x768] + b; q scaled by 1/sqrt(64).
        // wave owns M-tile (wv&1); A fragments cached in 64 VGPRs; 12 N-tiles.
        {
            const int mt = wv & 1;
            v16h afr[8];
#pragma unroll
            for (int kk = 0; kk < 8; ++kk)
                afr[kk] = lt_frag_a(aS + mt * 16 * 256 + kk * 32, 256, lane);
            for (int nt = (wv >> 1); nt < 48; nt += 4) {
                const int n0 = nt * 16;
                const float bias = bqkv[l * 768 + n0 + (lane & 15)];
                v8f acc;
#pragma unroll
                for (int r = 0; r < 8; ++r) acc[r] = bias;
                const _Float16* wr = Wqkv + (size_t)n0 * 256;
                __builtin_prefetch(wr + 4 * 16 * 256, 0, 1);   // next tile this wave
#pragma unroll
                for (int kk = 0; kk < 8; ++kk)
                    acc = lt_wmma(afr[kk], lt_frag_b(wr + kk * 32, 256, lane), acc);
                if (n0 < 256) {                  // q: row-major, *1/sqrt(hd)
                    const int col = n0 + (lane & 15);
#pragma unroll
                    for (int r = 0; r < 8; ++r)
                        qS[(mt * 16 + r + hi8) * 256 + col] = (_Float16)(acc[r] * 0.125f);
                } else if (n0 < 512) {           // k: row-major
                    const int col = n0 - 256 + (lane & 15);
#pragma unroll
                    for (int r = 0; r < 8; ++r)
                        kS[(mt * 16 + r + hi8) * 256 + col] = (_Float16)acc[r];
                } else {                         // v: transposed [chan][token]
                    const int c = n0 - 512 + (lane & 15);
                    const int mbase = mt * 16 + hi8;
#pragma unroll
                    for (int r = 0; r < 8; ++r)
                        vT[c * 32 + mbase + r] = (_Float16)acc[r];
                }
            }
        }
        __syncthreads();

        // ---- scores = q @ k^T per head (2 waves/head, K=64) ----
        {
            const int h = wv >> 1;
            const int t0 = (wv & 1) * 2;
            for (int t2 = t0; t2 < t0 + 2; ++t2) {
                const int mt = t2 & 1, nt = t2 >> 1;
                v8f acc;
#pragma unroll
                for (int r = 0; r < 8; ++r) acc[r] = 0.0f;
#pragma unroll
                for (int kk = 0; kk < 64; kk += 32)
                    acc = lt_wmma(lt_frag_a(qS + mt * 16 * 256 + h * 64 + kk, 256, lane),
                                  lt_frag_b(kS + nt * 16 * 256 + h * 64 + kk, 256, lane), acc);
                float* dst = attF + h * 1024;
                const int col = nt * 16 + (lane & 15);
#pragma unroll
                for (int r = 0; r < 8; ++r)
                    dst[(mt * 16 + r + hi8) * 32 + col] = acc[r];
            }
        }
        __syncthreads();

        // ---- softmax rows (128 rows total) -> attH f16 ----
        if (tid < 128) {
            const int h = tid >> 5, r = tid & 31;
            float* sr = attF + h * 1024 + r * 32;
            _Float16* dr = attH + h * 1024 + r * 32;
            float mx = -1e30f;
            for (int j = 0; j < 32; ++j) mx = fmaxf(mx, sr[j]);
            float sum = 0.0f;
            for (int j = 0; j < 32; ++j) { float e = __expf(sr[j] - mx); sr[j] = e; sum += e; }
            const float inv = 1.0f / sum;
            for (int j = 0; j < 32; ++j) dr[j] = (_Float16)(sr[j] * inv);
        }
        __syncthreads();

        // ---- o = att @ v -> aS (K=32; vT gives contiguous B loads) ----
        for (int tt = wv; tt < 32; tt += 8) {
            const int h = tt >> 3, mt = (tt >> 2) & 1, nt = tt & 3;
            v8f acc;
#pragma unroll
            for (int r = 0; r < 8; ++r) acc[r] = 0.0f;
            acc = lt_wmma(lt_frag_a(attH + h * 1024 + mt * 16 * 32, 32, lane),
                          lt_frag_b(vT + (h * 64 + nt * 16) * 32, 32, lane), acc);
            const int col = h * 64 + nt * 16 + (lane & 15);
#pragma unroll
            for (int r = 0; r < 8; ++r)
                aS[(mt * 16 + r + hi8) * 256 + col] = (_Float16)acc[r];
        }
        __syncthreads();

        // ---- out-proj + residual: x += o @ Wo^T + bo (A cached) ----
        {
            const int mt = wv & 1;
            v16h afr[8];
#pragma unroll
            for (int kk = 0; kk < 8; ++kk)
                afr[kk] = lt_frag_a(aS + mt * 16 * 256 + kk * 32, 256, lane);
            for (int nt = (wv >> 1); nt < 16; nt += 4) {
                const int n0 = nt * 16;
                const float bias = bo[l * 256 + n0 + (lane & 15)];
                v8f acc;
#pragma unroll
                for (int r = 0; r < 8; ++r) acc[r] = bias;
                const _Float16* wr = Wo + (size_t)n0 * 256;
                __builtin_prefetch(wr + 4 * 16 * 256, 0, 1);
#pragma unroll
                for (int kk = 0; kk < 8; ++kk)
                    acc = lt_wmma(afr[kk], lt_frag_b(wr + kk * 32, 256, lane), acc);
                const int col = n0 + (lane & 15);
#pragma unroll
                for (int r = 0; r < 8; ++r)
                    xS[(mt * 16 + r + hi8) * 256 + col] += acc[r];
            }
        }
        __syncthreads();

        // ---- LN2 -> aS ----
        lt_ln(xS, aS, ln2s + l * CC, ln2b + l * CC, tid);
        __syncthreads();

        // ---- FFN fc1 + ReLU -> hS [32x512] (A cached) ----
        {
            const int mt = wv & 1;
            v16h afr[8];
#pragma unroll
            for (int kk = 0; kk < 8; ++kk)
                afr[kk] = lt_frag_a(aS + mt * 16 * 256 + kk * 32, 256, lane);
            for (int nt = (wv >> 1); nt < 32; nt += 4) {
                const int n0 = nt * 16;
                const float bias = bfc1[l * 512 + n0 + (lane & 15)];
                v8f acc;
#pragma unroll
                for (int r = 0; r < 8; ++r) acc[r] = bias;
                const _Float16* wr = Wf1 + (size_t)n0 * 256;
                __builtin_prefetch(wr + 4 * 16 * 256, 0, 1);
#pragma unroll
                for (int kk = 0; kk < 8; ++kk)
                    acc = lt_wmma(afr[kk], lt_frag_b(wr + kk * 32, 256, lane), acc);
                const int col = n0 + (lane & 15);
#pragma unroll
                for (int r = 0; r < 8; ++r) {
                    const int m = mt * 16 + r + hi8;
                    hS[m * 512 + col] = (_Float16)fmaxf(acc[r], 0.0f);
                }
            }
        }
        __syncthreads();

        // ---- FFN fc2 + residual: x += h @ Wf2^T + b2 (K=512) ----
        for (int tt = wv; tt < 32; tt += 8) {
            const int mt = tt & 1, nt = tt >> 1, n0 = nt * 16;
            const float bias = bfc2[l * 256 + n0 + (lane & 15)];
            v8f acc;
#pragma unroll
            for (int r = 0; r < 8; ++r) acc[r] = bias;
            const _Float16* wr = Wf2 + (size_t)n0 * 512;
            __builtin_prefetch(wr + 8 * 16 * 512, 0, 1);
#pragma unroll
            for (int kk = 0; kk < 512; kk += 32)
                acc = lt_wmma(lt_frag_a(hS + mt * 16 * 512 + kk, 512, lane),
                              lt_frag_b(wr + kk, 512, lane), acc);
            const int col = n0 + (lane & 15);
#pragma unroll
            for (int r = 0; r < 8; ++r)
                xS[(mt * 16 + r + hi8) * 256 + col] += acc[r];
        }
        __syncthreads();
    }

    {
        const float4* src4 = (const float4*)xS;
        float4* dst4 = (float4*)xg;
        for (int i = tid; i < (NSS * CC) / 4; i += 256) dst4[i] = src4[i];
    }
}

// ---------------------------------------------------------------------------
// Scatter-back: first occurrence of each idx wins.
// ---------------------------------------------------------------------------
__global__ void lt_init_first_kernel(int* __restrict__ first) {
    int i = blockIdx.x * 256 + threadIdx.x;
    if (i < BB * NN) first[i] = 0x7FFFFFFF;
}

__global__ void lt_scatter_min_kernel(const int* __restrict__ idxbuf,
                                      int* __restrict__ first) {
    int t = blockIdx.x * 256 + threadIdx.x;
    if (t >= BB * NPP * NSS) return;
    const int b = t / (NPP * NSS);
    const int pos = t % (NPP * NSS);
    const int n = idxbuf[t];
    atomicMin(&first[b * NN + n], pos);
}

__global__ __launch_bounds__(256)
void lt_write_out_kernel(const int* __restrict__ first, const float* __restrict__ xbuf,
                         const float* __restrict__ feat, float* __restrict__ out) {
    const int bn = blockIdx.x;
    const int b = bn / NN, n = bn % NN;
    const int c = threadIdx.x;
    const int pos = first[bn];
    float v;
    if (pos < LF) {
        const int p = pos >> 5, s = pos & 31;
        v = xbuf[(((size_t)(b * NPP + p)) * NSS + s) * CC + c];
    } else {
        v = feat[((size_t)b * CC + c) * NN + n];
    }
    out[(size_t)bn * CC + c] = v;
}

// ---------------------------------------------------------------------------
// Host launcher
// ---------------------------------------------------------------------------
extern "C" void kernel_launch(void* const* d_in, const int* in_sizes, int n_in,
                              void* d_out, int out_size, void* d_ws, size_t ws_size,
                              hipStream_t stream) {
    (void)in_sizes; (void)n_in; (void)out_size; (void)ws_size;

    const float* xyz   = (const float*)d_in[0];
    const float* feat  = (const float*)d_in[1];
    const float* pe1w  = (const float*)d_in[2];
    const float* pe1b  = (const float*)d_in[3];
    const float* bng   = (const float*)d_in[4];
    const float* bnb   = (const float*)d_in[5];
    const float* bnm   = (const float*)d_in[6];
    const float* bnv   = (const float*)d_in[7];
    const float* pe2w  = (const float*)d_in[8];
    const float* pe2b  = (const float*)d_in[9];
    const float* ipw   = (const float*)d_in[10];
    const float* ipb   = (const float*)d_in[11];
    const float* opw   = (const float*)d_in[12];
    const float* opb   = (const float*)d_in[13];
    const float* ln1s  = (const float*)d_in[14];
    const float* ln1b  = (const float*)d_in[15];
    const float* ln2s  = (const float*)d_in[16];
    const float* ln2b  = (const float*)d_in[17];
    const float* fc1w  = (const float*)d_in[18];
    const float* fc1b  = (const float*)d_in[19];
    const float* fc2w  = (const float*)d_in[20];
    const float* fc2b  = (const float*)d_in[21];
    float* out = (float*)d_out;

    // Workspace carve-up
    char* ws = (char*)d_ws;
    size_t off = 0;
    auto carve = [&](size_t bytes) -> void* {
        void* p = ws + off;
        off += bytes;
        off = (off + 255) & ~(size_t)255;
        return p;
    };
    float*    xbuf   = (float*)carve((size_t)GG * NSS * CC * sizeof(float));    // 134 MB
    _Float16* wqkv16 = (_Float16*)carve((size_t)LL * 768 * 256 * sizeof(_Float16));
    _Float16* wo16   = (_Float16*)carve((size_t)LL * 256 * 256 * sizeof(_Float16));
    _Float16* wf1_16 = (_Float16*)carve((size_t)LL * 512 * 256 * sizeof(_Float16));
    _Float16* wf2_16 = (_Float16*)carve((size_t)LL * 256 * 512 * sizeof(_Float16));
    int*      fpsidx = (int*)carve((size_t)BB * NPP * sizeof(int));
    int*      idxbuf = (int*)carve((size_t)BB * NPP * NSS * sizeof(int));
    float*    dists  = (float*)carve((size_t)BB * NN * sizeof(float));
    int*      first  = (int*)carve((size_t)BB * NN * sizeof(int));

    // 0) weights -> f16 (hot in L2 for all 4096 groups)
    {
        int n;
        n = LL * 768 * 256; lt_cvt_f16_kernel<<<(n + 255) / 256, 256, 0, stream>>>(ipw,  wqkv16, n);
        n = LL * 256 * 256; lt_cvt_f16_kernel<<<(n + 255) / 256, 256, 0, stream>>>(opw,  wo16,   n);
        n = LL * 512 * 256; lt_cvt_f16_kernel<<<(n + 255) / 256, 256, 0, stream>>>(fc1w, wf1_16, n);
        n = LL * 256 * 512; lt_cvt_f16_kernel<<<(n + 255) / 256, 256, 0, stream>>>(fc2w, wf2_16, n);
    }

    // 1) FPS (one block per batch)
    lt_fps_kernel<<<BB, 256, 0, stream>>>(xyz, dists, fpsidx);

    // 2) Ball query
    lt_ball_query_kernel<<<(GG + 255) / 256, 256, 0, stream>>>(xyz, fpsidx, idxbuf);

    // 3) Gather + positional encoding -> x0
    lt_pe_gather_kernel<<<GG, 256, 0, stream>>>(xyz, feat, pe1w, pe1b, bng, bnb,
                                                bnm, bnv, pe2w, pe2b,
                                                fpsidx, idxbuf, xbuf);

    // 4) Fused 2-layer transformer, 120 KB dynamic LDS (2 groups / 320KB WGP)
    const int smem_bytes = 120 * 1024;
    hipFuncSetAttribute(reinterpret_cast<const void*>(lt_transformer_kernel),
                        hipFuncAttributeMaxDynamicSharedMemorySize, smem_bytes);
    lt_transformer_kernel<<<GG, 256, smem_bytes, stream>>>(
        xbuf, wqkv16, ipb, wo16, opb, ln1s, ln1b, ln2s, ln2b,
        wf1_16, fc1b, wf2_16, fc2b);

    // 5) Scatter-back (first occurrence wins)
    lt_init_first_kernel<<<(BB * NN + 255) / 256, 256, 0, stream>>>(first);
    lt_scatter_min_kernel<<<(BB * NPP * NSS + 255) / 256, 256, 0, stream>>>(idxbuf, first);
    lt_write_out_kernel<<<BB * NN, 256, 0, stream>>>(first, xbuf, feat, out);
}
